// LePEAttention_23115513987240
// MI455X (gfx1250) — matrix operand order repaired and checked by reading, not compiled, not run on hardware
//
#include <hip/hip_runtime.h>
#include <hip/hip_bf16.h>

typedef __attribute__((ext_vector_type(16))) _Float16 v16h;
typedef __attribute__((ext_vector_type(8)))  float    v8f;

// Problem constants (fixed by the reference's setup_inputs)
#define BATCH   4
#define H_IMG   112
#define W_IMG   112
#define L_TOK   (H_IMG * W_IMG)   // 12544
#define CH      128
#define NH      4
#define HD      32                 // head dim
#define WSP     7                  // split size (stripe width)
#define WIN     (H_IMG * WSP)      // 784 tokens per window
#define NT      (WIN / 16)         // 49 16-row tiles
#define NWIN    (W_IMG / WSP)      // 16 windows per image
#define PADW    800                // WIN padded to multiple of 32
#define NCHUNK  (PADW / 32)        // 25 key chunks of 32

// ---------------------------------------------------------------------------
// Fused windowed attention: one wave per (window, head, query-tile).
// Pass 1: S = (Q*scale) K^T via v_wmma_f32_16x16x32_f16  -> LDS (f32)
// Softmax from LDS (vectorized b128 LDS traffic, deferred normalization).
// Pass 2: X = P V via v_wmma_f32_16x16x32_f16 (two N-halves), store to out.
// ---------------------------------------------------------------------------
__global__ __launch_bounds__(32)
void lepe_attn_kernel(const float* __restrict__ qkv, float* __restrict__ out) {
    const int qt   = blockIdx.x;        // query tile 0..48
    const int h    = blockIdx.y;        // head 0..3
    const int wi   = blockIdx.z;        // b*16 + wx
    const int b    = wi >> 4;
    const int wx   = wi & 15;
    const int lane = threadIdx.x;       // 0..31 (wave32)
    const int half = lane >> 4;         // 0 or 1
    const int ln   = lane & 15;

    __shared__ float sc[16][PADW];      // score / prob tile (f32), rows 16B-aligned
    __shared__ float rowsum[16];

    const float scale = 0.17677669529663687f;  // 32^-0.5

    const float* qp = qkv;                                   // [B, L, C]
    const float* kp = qkv + (size_t)1 * BATCH * L_TOK * CH;
    const float* vp = qkv + (size_t)2 * BATCH * L_TOK * CH;

    // window token t (0..783) -> flat image token index l
    auto tok2l = [&](int t) -> int {
        int yy = t / WSP, xx = t - yy * WSP;
        return yy * W_IMG + wx * WSP + xx;
    };

    // ---- Q A-fragment (16x32 f16, pre-scaled) -----------------------------
    // A operand, contiguous k order used here: lane M = ln; lanes<16 hold K 0-15;
    // lanes>=16 hold K 16-31.
    v16h aq;
    {
        int t = qt * 16 + ln;
        int l = tok2l(t);
        const float* qrow = qp + ((size_t)b * L_TOK + l) * CH + h * HD;
        const int kb = half * 16;
#pragma unroll
        for (int i = 0; i < 8; ++i)
            aq[i] = (_Float16)(qrow[kb + i] * scale);
#pragma unroll
        for (int i = 8; i < 16; ++i)
            aq[i] = (_Float16)(qrow[kb + i] * scale);   // K = kb+8..kb+15
    }

    // ---- Pass 1: scores to LDS --------------------------------------------
    // B operand, same contiguous k order as A: lane N = ln; K = half*16 + i.
    for (int kt = 0; kt < NT; ++kt) {
        int tk = kt * 16 + ln;
        int lk = tok2l(tk);
        const float* krow = kp + ((size_t)b * L_TOK + lk) * CH + h * HD;
        // prefetch next tile's row while this tile computes
        if (kt + 1 < NT) {
            int tn = kt * 16 + 16 + ln;
            __builtin_prefetch(kp + ((size_t)b * L_TOK + tok2l(tn)) * CH + h * HD, 0, 1);
        }
        const int kb = half * 16;
        v16h bk;
#pragma unroll
        for (int i = 0; i < 16; ++i)
            bk[i] = (_Float16)krow[kb + i];
        v8f c = {};
        c = __builtin_amdgcn_wmma_f32_16x16x32_f16(
                false, aq, false, bk, (short)0, c, false, false);
        // C/D layout: VGPR r -> row (r or r+8), col = ln
#pragma unroll
        for (int r = 0; r < 8; ++r)
            sc[r + half * 8][kt * 16 + ln] = c[r];
    }
    // pad columns 784..799 so exp() gives exactly 0
#pragma unroll
    for (int r = 0; r < 8; ++r)
        sc[r + half * 8][WIN + ln] = -1e30f;
    __syncthreads();

    // ---- Softmax (unnormalized; keep 1/rowsum for epilogue) ---------------
    // Lane handles row=ln, column half `half` (400 floats = 100 float4s).
    {
        const int row = ln;
        float4* rowp = (float4*)&sc[row][half * (PADW / 2)];
        float m = -1e30f;
        for (int j = 0; j < PADW / 8; ++j) {
            float4 v = rowp[j];
            m = fmaxf(m, fmaxf(fmaxf(v.x, v.y), fmaxf(v.z, v.w)));
        }
        m = fmaxf(m, __shfl_xor(m, 16, 32));
        float s = 0.f;
        for (int j = 0; j < PADW / 8; ++j) {
            float4 v = rowp[j];
            v.x = __expf(v.x - m);
            v.y = __expf(v.y - m);
            v.z = __expf(v.z - m);
            v.w = __expf(v.w - m);
            rowp[j] = v;
            s += (v.x + v.y) + (v.z + v.w);
        }
        s += __shfl_xor(s, 16, 32);
        if (half == 0) rowsum[row] = s;
    }
    __syncthreads();

    // ---- Pass 2: X = P V ---------------------------------------------------
    v8f acc0 = {}, acc1 = {};   // hd 0-15 and 16-31
    for (int chn = 0; chn < NCHUNK; ++chn) {
        // P A-fragment from LDS (same A layout as Q, K-dim = keys)
        v16h ap;
        {
            const int base = chn * 32 + half * 16;
#pragma unroll
            for (int i = 0; i < 8; ++i)
                ap[i] = (_Float16)sc[ln][base + i];
#pragma unroll
            for (int i = 8; i < 16; ++i)
                ap[i] = (_Float16)sc[ln][base + i];
        }
        // V B-fragments: lane N = ln (hd col), K = key = chn*32 + half*16 + i
        const int kb = chn * 32 + half * 16;
        v16h bv0, bv1;
#pragma unroll
        for (int i = 0; i < 16; ++i) {
            int key = kb + i;
            if (key > WIN - 1) key = WIN - 1;   // P==0 there, value irrelevant
            int lk = tok2l(key);
            const float* vrow = vp + ((size_t)b * L_TOK + lk) * CH + h * HD;
            bv0[i] = (_Float16)vrow[ln];
            bv1[i] = (_Float16)vrow[ln + 16];
        }
        acc0 = __builtin_amdgcn_wmma_f32_16x16x32_f16(
                   false, ap, false, bv0, (short)0, acc0, false, false);
        acc1 = __builtin_amdgcn_wmma_f32_16x16x32_f16(
                   false, ap, false, bv1, (short)0, acc1, false, false);
    }

    // ---- Epilogue: normalize and store ------------------------------------
#pragma unroll
    for (int r = 0; r < 8; ++r) {
        const int M   = r + half * 8;
        const float inv = 1.0f / rowsum[M];
        const int t = qt * 16 + M;
        const int l = tok2l(t);
        float* orow = out + ((size_t)b * L_TOK + l) * CH + h * HD;
        orow[ln]      = acc0[r] * inv;
        orow[ln + 16] = acc1[r] * inv;
    }
}

// ---------------------------------------------------------------------------
// LePE: depthwise 3x3 conv on V arranged per-window (zero padding at the
// window boundary, i.e. stripe edges), added into out. Runs after the
// attention kernel on the same stream.
// ---------------------------------------------------------------------------
__global__ __launch_bounds__(256)
void lepe_conv_add_kernel(const float* __restrict__ qkv,
                          const float* __restrict__ cw,
                          const float* __restrict__ cb,
                          float* __restrict__ out) {
    const int idx = blockIdx.x * blockDim.x + threadIdx.x;
    const int total = BATCH * L_TOK * CH;
    if (idx >= total) return;
    const int c = idx & (CH - 1);
    const int l = (idx / CH) % L_TOK;
    const int b = idx / (CH * L_TOK);
    const int y = l / W_IMG;
    const int x = l - y * W_IMG;
    const int x0 = (x / WSP) * WSP;          // left edge of this stripe window

    const float* vp = qkv + (size_t)2 * BATCH * L_TOK * CH;
    float acc = cb[c];
#pragma unroll
    for (int ky = 0; ky < 3; ++ky) {
        const int yy = y + ky - 1;
        if (yy < 0 || yy >= H_IMG) continue;
#pragma unroll
        for (int kx = 0; kx < 3; ++kx) {
            const int xx = x + kx - 1;
            if (xx < x0 || xx >= x0 + WSP) continue;   // window-edge zero pad
            acc += cw[c * 9 + ky * 3 + kx] *
                   vp[((size_t)b * L_TOK + yy * W_IMG + xx) * CH + c];
        }
    }
    out[idx] += acc;
}

// ---------------------------------------------------------------------------
extern "C" void kernel_launch(void* const* d_in, const int* in_sizes, int n_in,
                              void* d_out, int out_size, void* d_ws, size_t ws_size,
                              hipStream_t stream) {
    const float* qkv = (const float*)d_in[0];   // [3, B, L, C] f32
    const float* cw  = (const float*)d_in[1];   // [C, 1, 3, 3] f32
    const float* cb  = (const float*)d_in[2];   // [C] f32
    float* out = (float*)d_out;                 // [B, L, C] f32

    dim3 grid(NT, NH, BATCH * NWIN);            // 49 x 4 x 64 waves
    lepe_attn_kernel<<<grid, 32, 0, stream>>>(qkv, out);

    const int total = BATCH * L_TOK * CH;
    lepe_conv_add_kernel<<<(total + 255) / 256, 256, 0, stream>>>(qkv, cw, cb, out);
}
